// ParallelEncoderBlock_20091857011462
// MI455X (gfx1250) — matrix operand order repaired and checked by reading, not compiled, not run on hardware
//
#include <hip/hip_runtime.h>
#include <hip/hip_bf16.h>
#include <stdint.h>

// ---------------- types ----------------
typedef __attribute__((ext_vector_type(16))) __bf16 v16bf;
typedef __attribute__((ext_vector_type(8)))  float  v8f;

struct FragBF { union { v16bf v; uint32_t u[8]; }; };
struct FragF  { union { v8f  v; float    f[8]; }; };

__device__ __forceinline__ unsigned short f2bf(float f) {
    union { float f; uint32_t u; } c; c.f = f;
    uint32_t u = c.u;
    uint32_t r = (u + 0x7FFFu + ((u >> 16) & 1u)) >> 16;   // RNE
    return (unsigned short)r;
}
__device__ __forceinline__ float bf2f(unsigned short h) {
    union { uint32_t u; float f; } c; c.u = ((uint32_t)h) << 16;
    return c.f;
}
// A-fragment K base for 16-bit A 16x32 (contiguous k = 16*hf + s): element pair v, wave half hf
__device__ __forceinline__ int a_kbase(int v, int hf) {
    return 16 * hf + 2 * v;
}

// ---- CDNA5 async memory->LDS copy (ASYNCcnt path, ISA §10/§15.18.3) ----
// LDS byte address = low 32 bits of the generic shared pointer (LDS aperture
// truncates to addr[31:0], ISA 10.2).
__device__ __forceinline__ void async_copy_b128(const void* gptr, void* lptr) {
    uint32_t lds = (uint32_t)(size_t)lptr;
    asm volatile("global_load_async_to_lds_b128 %0, %1, off"
                 :: "v"(lds), "v"(gptr) : "memory");
}
__device__ __forceinline__ void wait_async0() {
    asm volatile("s_wait_asynccnt 0x0" ::: "memory");
}

#define MODEL_DIM 1024
#define FF_DIM    4096
#define SEQ       2048
#define NHEADS    16
#define HDIM      64
#define PROJ_N    11264      // 3*1024 + 2*4096
#define A2_N      5120       // 1024 (attn concat) + 4096 (gated ff)
#define NTOK      4096       // B*S

// ---------------- weight conversion ----------------
__global__ void conv_f32_bf16(const float* __restrict__ src,
                              unsigned short* __restrict__ dst, int n) {
    for (int i = blockIdx.x * blockDim.x + threadIdx.x; i < n;
         i += gridDim.x * blockDim.x)
        dst[i] = f2bf(src[i]);
}

// stack w_attn (1024x1024) on top of w_ff (4096x1024) -> W2 (5120x1024) bf16
__global__ void conv_w2(const float* __restrict__ w_attn,
                        const float* __restrict__ w_ff,
                        unsigned short* __restrict__ W2) {
    int n = A2_N * MODEL_DIM;
    for (int i = blockIdx.x * blockDim.x + threadIdx.x; i < n;
         i += gridDim.x * blockDim.x) {
        int k = i / MODEL_DIM, c = i % MODEL_DIM;
        float v = (k < MODEL_DIM) ? w_attn[k * MODEL_DIM + c]
                                  : w_ff[(k - MODEL_DIM) * MODEL_DIM + c];
        W2[i] = f2bf(v);
    }
}

// ---------------- layernorm (one row / block) ----------------
__global__ __launch_bounds__(256)
void layernorm_kernel(const float* __restrict__ x,
                      const float* __restrict__ w,
                      const float* __restrict__ b,
                      unsigned short* __restrict__ y) {
    __shared__ float red[256];
    int row = blockIdx.x, tid = threadIdx.x;
    const float* xr = x + (size_t)row * MODEL_DIM;
    float lx[4];
    float s = 0.f;
    for (int i = 0; i < 4; ++i) { lx[i] = xr[tid + i * 256]; s += lx[i]; }
    red[tid] = s; __syncthreads();
    for (int st = 128; st > 0; st >>= 1) {
        if (tid < st) red[tid] += red[tid + st];
        __syncthreads();
    }
    float mean = red[0] * (1.0f / MODEL_DIM);
    __syncthreads();
    float sq = 0.f;
    for (int i = 0; i < 4; ++i) { float d = lx[i] - mean; sq += d * d; }
    red[tid] = sq; __syncthreads();
    for (int st = 128; st > 0; st >>= 1) {
        if (tid < st) red[tid] += red[tid + st];
        __syncthreads();
    }
    float rstd = rsqrtf(red[0] * (1.0f / MODEL_DIM) + 1e-5f);
    for (int i = 0; i < 4; ++i) {
        int c = tid + i * 256;
        y[(size_t)row * MODEL_DIM + c] = f2bf((lx[i] - mean) * rstd * w[c] + b[c]);
    }
}

// ---------------- generic bf16 WMMA GEMM ----------------
// C[M,N] = A[M,K] * B[K,N]; A,B bf16 row-major.
// WB==1: store bf16 into Cb. WB==0: Cf[idx] = resid[idx] + acc (fp32).
// Tiling: WG 256 thr = 8 waves; block tile 128x64; wave tile 32x32 (2x2 WMMA).
// A tile staged memory->LDS via async copy (ASYNCcnt); B tile staged with a
// transpose on the way in (manual path).
template <int WB>
__global__ __launch_bounds__(256)
void gemm_bf16_kernel(const unsigned short* __restrict__ A, int lda,
                      const unsigned short* __restrict__ B, int ldb,
                      int M, int N, int K,
                      unsigned short* __restrict__ Cb,
                      float* __restrict__ Cf,
                      const float* __restrict__ resid) {
    __shared__ unsigned short sA[128 * 32];   // [m][k]
    __shared__ unsigned short sB[64 * 32];    // [n][k]  (transposed)
    const int tid  = threadIdx.x;
    const int wave = tid >> 5, lane = tid & 31;
    const int lm = lane & 15, hf = lane >> 4;
    const int m0 = blockIdx.y * 128;
    const int n0 = blockIdx.x * 64;
    const int wm = (wave & 3) * 32;           // 4 waves along M
    const int wn = (wave >> 2) * 32;          // 2 waves along N

    FragF acc[2][2];
    for (int i = 0; i < 2; ++i)
        for (int j = 0; j < 2; ++j)
            for (int r = 0; r < 8; ++r) acc[i][j].f[r] = 0.f;

    for (int k0 = 0; k0 < K; k0 += 32) {
        // stage A tile 128x32 bf16: 512 16B chunks, 2 async copies per thread
        for (int i = 0; i < 2; ++i) {
            int idx = tid + i * 256;
            int row = idx >> 2, chunk = idx & 3;
            async_copy_b128(&A[(size_t)(m0 + row) * lda + k0 + chunk * 8],
                            &sA[row * 32 + chunk * 8]);
        }
        // stage B tile transposed: 32x64 -> sB[n][k]
        for (int i = 0; i < 8; ++i) {
            int idx = tid + i * 256;
            int k = idx >> 6, n = idx & 63;
            sB[n * 32 + k] = B[(size_t)(k0 + k) * ldb + n0 + n];
        }
        if (k0 + 32 < K) {
            __builtin_prefetch(&A[(size_t)(m0 + (tid >> 2)) * lda + k0 + 32], 0, 1);
            __builtin_prefetch(&B[(size_t)(k0 + 32 + (tid >> 6)) * ldb + n0 + (tid & 63)], 0, 1);
        }
        wait_async0();
        __syncthreads();

        FragBF a[2], b[2];
        for (int i = 0; i < 2; ++i) {
            int row = wm + i * 16 + lm;
            for (int v = 0; v < 8; ++v)
                a[i].u[v] = *(const uint32_t*)&sA[row * 32 + a_kbase(v, hf)];
        }
        for (int j = 0; j < 2; ++j) {
            int n = wn + j * 16 + lm;
            for (int v = 0; v < 8; ++v)
                b[j].u[v] = *(const uint32_t*)&sB[n * 32 + 16 * hf + v * 2];
        }
        for (int i = 0; i < 2; ++i)
            for (int j = 0; j < 2; ++j)
                acc[i][j].v = __builtin_amdgcn_wmma_f32_16x16x32_bf16(
                    false, a[i].v, false, b[j].v, (short)0, acc[i][j].v,
                    false, false);
        __syncthreads();
    }

    // epilogue (C layout: lane holds n = lane%16, VGPR r -> m = r + 8*(lane/16))
    for (int i = 0; i < 2; ++i)
        for (int j = 0; j < 2; ++j)
            for (int r = 0; r < 8; ++r) {
                int row = m0 + wm + i * 16 + r + 8 * hf;
                int col = n0 + wn + j * 16 + lm;
                size_t idx = (size_t)row * N + col;
                if (WB) Cb[idx] = f2bf(acc[i][j].f[r]);
                else    Cf[idx] = resid[idx] + acc[i][j].f[r];
            }
}

// ---------------- gated gelu: A2[:,1024+j] = ff * gelu(gate) ----------------
__global__ void gelu_mul_kernel(const unsigned short* __restrict__ proj,
                                unsigned short* __restrict__ A2) {
    const int n = NTOK * FF_DIM;
    for (int i = blockIdx.x * blockDim.x + threadIdx.x; i < n;
         i += gridDim.x * blockDim.x) {
        int row = i / FF_DIM, j = i % FF_DIM;
        float ff = bf2f(proj[(size_t)row * PROJ_N + 3 * MODEL_DIM + j]);
        float g  = bf2f(proj[(size_t)row * PROJ_N + 3 * MODEL_DIM + FF_DIM + j]);
        float gl = 0.5f * g *
                   (1.0f + tanhf(0.7978845608028654f * (g + 0.044715f * g * g * g)));
        A2[(size_t)row * A2_N + MODEL_DIM + j] = f2bf(ff * gl);
    }
}

// ---------------- flash attention ----------------
// grid (S/64, B*H); 128 thr = 4 waves, each wave owns 16 query rows.
// K tile staged via async memory->LDS copies; V tile staged transposed.
__global__ __launch_bounds__(128)
void attn_kernel(const unsigned short* __restrict__ proj,
                 unsigned short* __restrict__ A2) {
    __shared__ unsigned short sK[32 * 64];       // [key][d]
    __shared__ unsigned short sVt[64 * 32];      // [d][key] (transposed)
    __shared__ unsigned short sP[4][16 * 32];    // per-wave P tile [m][key]

    const int bh = blockIdx.y;
    const int b = bh >> 4, h = bh & 15;
    const int wave = threadIdx.x >> 5, lane = threadIdx.x & 31;
    const int lm = lane & 15, hf = lane >> 4;
    const int qrow0 = b * SEQ + blockIdx.x * 64 + wave * 16;  // token row of m=0
    const size_t LDP = PROJ_N;

    // Q fragments (A-matrix, two K-chunks of 32 along d)
    FragBF qf[2];
    for (int ks = 0; ks < 2; ++ks)
        for (int v = 0; v < 8; ++v)
            qf[ks].u[v] = *(const uint32_t*)&proj[(size_t)(qrow0 + lm) * LDP +
                                                  h * HDIM + ks * 32 + a_kbase(v, hf)];

    float run_m[8], run_l[8];
    FragF o[4];
    for (int r = 0; r < 8; ++r) { run_m[r] = -1e30f; run_l[r] = 0.f; }
    for (int j = 0; j < 4; ++j)
        for (int r = 0; r < 8; ++r) o[j].f[r] = 0.f;

    for (int kt0 = 0; kt0 < SEQ; kt0 += 32) {
        // K tile 32x64 bf16: 256 16B chunks, 2 async copies per thread
        for (int i = 0; i < 2; ++i) {
            int c = threadIdx.x + i * 128;
            int kr = c >> 3, d8 = c & 7;                 // chunk of 8 d values
            async_copy_b128(&proj[(size_t)(b * SEQ + kt0 + kr) * LDP +
                                  MODEL_DIM + h * HDIM + d8 * 8],
                            &sK[kr * 64 + d8 * 8]);
        }
        // V tile transposed (needs per-element transpose on the way in)
        for (int i = 0; i < 8; ++i) {
            int idx = threadIdx.x + i * 128;             // 1024 u32
            int kr = idx >> 5, dc = idx & 31;            // u32 covers 2 d values
            uint32_t vv = *(const uint32_t*)&proj[(size_t)(b * SEQ + kt0 + kr) * LDP +
                                                  2 * MODEL_DIM + h * HDIM + dc * 2];
            sVt[(2 * dc) * 32 + kr]     = (unsigned short)(vv & 0xFFFFu);
            sVt[(2 * dc + 1) * 32 + kr] = (unsigned short)(vv >> 16);
        }
        wait_async0();
        __syncthreads();

        // S = Q K^T  (two key blocks of 16)
        FragF s0, s1;
        for (int r = 0; r < 8; ++r) { s0.f[r] = 0.f; s1.f[r] = 0.f; }
        for (int ks = 0; ks < 2; ++ks) {
            FragBF kb0, kb1;
            for (int v = 0; v < 8; ++v) {
                kb0.u[v] = *(const uint32_t*)&sK[lm * 64 + ks * 32 + 16 * hf + v * 2];
                kb1.u[v] = *(const uint32_t*)&sK[(16 + lm) * 64 + ks * 32 + 16 * hf + v * 2];
            }
            s0.v = __builtin_amdgcn_wmma_f32_16x16x32_bf16(false, qf[ks].v, false,
                                                           kb0.v, (short)0, s0.v, false, false);
            s1.v = __builtin_amdgcn_wmma_f32_16x16x32_bf16(false, qf[ks].v, false,
                                                           kb1.v, (short)0, s1.v, false, false);
        }

        // online softmax (row m = r + 8*hf lives in one 16-lane half)
        const float scale = 0.125f;   // 1/sqrt(64)
        float p0[8], p1[8];
        for (int r = 0; r < 8; ++r) {
            float x0 = s0.f[r] * scale, x1 = s1.f[r] * scale;
            float tm = fmaxf(x0, x1);
            for (int msk = 1; msk < 16; msk <<= 1)
                tm = fmaxf(tm, __shfl_xor(tm, msk, 32));
            float nm = fmaxf(run_m[r], tm);
            float corr = __expf(run_m[r] - nm);
            float e0 = __expf(x0 - nm), e1 = __expf(x1 - nm);
            float rs = e0 + e1;
            for (int msk = 1; msk < 16; msk <<= 1)
                rs += __shfl_xor(rs, msk, 32);
            run_l[r] = run_l[r] * corr + rs;
            run_m[r] = nm;
            p0[r] = e0; p1[r] = e1;
            for (int j = 0; j < 4; ++j) o[j].f[r] *= corr;
        }

        // re-layout P: C-frag -> LDS [m][key] -> A-frag
        for (int r = 0; r < 8; ++r) {
            sP[wave][(r + 8 * hf) * 32 + lm]      = f2bf(p0[r]);
            sP[wave][(r + 8 * hf) * 32 + 16 + lm] = f2bf(p1[r]);
        }
        __syncthreads();
        FragBF pa;
        for (int v = 0; v < 8; ++v)
            pa.u[v] = *(const uint32_t*)&sP[wave][lm * 32 + a_kbase(v, hf)];

        // O += P V  (4 output blocks of 16 along d)
        for (int j = 0; j < 4; ++j) {
            FragBF vb;
            for (int v = 0; v < 8; ++v)
                vb.u[v] = *(const uint32_t*)&sVt[(j * 16 + lm) * 32 + 16 * hf + v * 2];
            o[j].v = __builtin_amdgcn_wmma_f32_16x16x32_bf16(false, pa.v, false,
                                                             vb.v, (short)0, o[j].v, false, false);
        }
        __syncthreads();
    }

    // write concat: A2[token][h*64 + d]
    for (int j = 0; j < 4; ++j)
        for (int r = 0; r < 8; ++r) {
            int token = qrow0 + r + 8 * hf;
            int col = h * HDIM + j * 16 + lm;
            A2[(size_t)token * A2_N + col] = f2bf(o[j].f[r] / run_l[r]);
        }
}

// ---------------- host launch ----------------
extern "C" void kernel_launch(void* const* d_in, const int* in_sizes, int n_in,
                              void* d_out, int out_size, void* d_ws, size_t ws_size,
                              hipStream_t stream) {
    const float* inputs = (const float*)d_in[0];
    const float* ln_w   = (const float*)d_in[1];
    const float* ln_b   = (const float*)d_in[2];
    const float* w_in   = (const float*)d_in[3];
    const float* w_attn = (const float*)d_in[4];
    const float* w_ff   = (const float*)d_in[5];
    float* out = (float*)d_out;

    char* ws = (char*)d_ws;
    size_t off = 0;
    unsigned short* Xln  = (unsigned short*)(ws + off); off += (size_t)NTOK * MODEL_DIM * 2;   // 8 MB
    unsigned short* Wp   = (unsigned short*)(ws + off); off += (size_t)MODEL_DIM * PROJ_N * 2; // 23 MB
    unsigned short* W2   = (unsigned short*)(ws + off); off += (size_t)A2_N * MODEL_DIM * 2;   // 10.5 MB
    unsigned short* proj = (unsigned short*)(ws + off); off += (size_t)NTOK * PROJ_N * 2;      // 92 MB
    unsigned short* A2   = (unsigned short*)(ws + off); off += (size_t)NTOK * A2_N * 2;        // 42 MB

    conv_f32_bf16<<<1024, 256, 0, stream>>>(w_in, Wp, MODEL_DIM * PROJ_N);
    conv_w2<<<512, 256, 0, stream>>>(w_attn, w_ff, W2);
    layernorm_kernel<<<NTOK, 256, 0, stream>>>(inputs, ln_w, ln_b, Xln);

    // proj = Xln @ w_in  -> bf16
    gemm_bf16_kernel<1><<<dim3(PROJ_N / 64, NTOK / 128), 256, 0, stream>>>(
        Xln, MODEL_DIM, Wp, PROJ_N, NTOK, PROJ_N, MODEL_DIM, proj, nullptr, nullptr);

    gelu_mul_kernel<<<4096, 256, 0, stream>>>(proj, A2);
    attn_kernel<<<dim3(SEQ / 64, 2 * NHEADS), 128, 0, stream>>>(proj, A2);

    // out = inputs + A2 @ [w_attn; w_ff]
    gemm_bf16_kernel<0><<<dim3(MODEL_DIM / 64, NTOK / 128), 256, 0, stream>>>(
        A2, A2_N, W2, MODEL_DIM, NTOK, MODEL_DIM, A2_N, nullptr, out, inputs);
}
